// AttentionModule_16484084483034
// MI455X (gfx1250) — matrix operand order repaired and checked by reading, not compiled, not run on hardware
//
#include <hip/hip_runtime.h>
#include <hip/hip_bf16.h>

typedef __attribute__((ext_vector_type(16))) __bf16 v16bf;
typedef __attribute__((ext_vector_type(4)))  __bf16 v4bf;
typedef __attribute__((ext_vector_type(8)))  float  v8f;

#if defined(__has_builtin)
#if __has_builtin(__builtin_amdgcn_global_load_async_to_lds_b128) && \
    __has_builtin(__builtin_amdgcn_s_wait_asynccnt)
#define USE_ASYNC_LDS 1
#endif
#endif

#ifdef USE_ASYNC_LDS
typedef int v4i __attribute__((vector_size(16)));
typedef __attribute__((address_space(1))) v4i* g4p;   // global int4*
typedef __attribute__((address_space(3))) v4i* l4p;   // LDS int4*
#endif

static __device__ __forceinline__ __bf16 f2bf(float f) {
    unsigned int u = __builtin_bit_cast(unsigned int, f);
    unsigned int r = (u + 0x7FFFu + ((u >> 16) & 1u)) >> 16;
    unsigned short s = (unsigned short)r;
    return __builtin_bit_cast(__bf16, s);
}

// ---------------------------------------------------------------------------
// Mask length sums: lens[0..3] = sum(x_mask[b]), lens[4..7] = sum(context_mask[b])
// ---------------------------------------------------------------------------
__global__ __launch_bounds__(256) void lens_kernel(const float* __restrict__ xm,
                                                   const float* __restrict__ cm,
                                                   float* __restrict__ lens) {
    __shared__ float red[256];
    int which = blockIdx.x >> 2;
    int b = blockIdx.x & 3;
    const float* src = (which == 0 ? xm : cm) + b * 1024;
    int t = threadIdx.x;
    float v = src[t] + src[t + 256] + src[t + 512] + src[t + 768];
    red[t] = v;
    __syncthreads();
    for (int s = 128; s > 0; s >>= 1) {
        if (t < s) red[t] += red[t + s];
        __syncthreads();
    }
    if (t == 0) lens[which * 4 + b] = red[0];
}

// ---------------------------------------------------------------------------
// GEMM core: C[4096 x 1024] = A[4096 x 1024] @ W^T,  W is [1024 x 1024] row-major.
// MODE 0: A = x (read transposed, fp32), epilogue bias + RoPE(len_q) -> Q bf16
// MODE 1: A = context (fp32),            epilogue bias + RoPE(len_k) -> K bf16
// MODE 2: A = context (fp32),            epilogue bias               -> V bf16
// MODE 3: A = attn-out (bf16),           epilogue bias + x_mask + transpose -> fp32 out
// Block: 128 threads (4 waves), 64x64 output tile; each wave: 16 rows x 64 cols.
// LDS: As row-major [64][40] (80B stride), Bt operand-major [n][k] [64][40]
//      so every WMMA fragment gather is 2x aligned ds_load_b128.
// ---------------------------------------------------------------------------
template <int MODE>
__global__ __launch_bounds__(128) void gemm_kernel(
    const float* __restrict__ Afp, const __bf16* __restrict__ Abf,
    const float* __restrict__ W,   const float* __restrict__ bias,
    const float* __restrict__ lens, const float* __restrict__ xmask,
    __bf16* __restrict__ outb, float* __restrict__ outf) {
    __shared__ alignas(16) __bf16 As[64][40];
    __shared__ alignas(16) __bf16 Bt[64][40];

    const int tid  = threadIdx.x;
    const int wave = tid >> 5;
    const int lane = tid & 31;
    const int r    = lane & 15;
    const int half = lane >> 4;
    const int m0 = blockIdx.x * 64;
    const int n0 = blockIdx.y * 64;

    v8f acc[4] = {};

    for (int k0 = 0; k0 < 1024; k0 += 32) {
        if constexpr (MODE == 0) {
            // A[m][k] = x[b, k, t]; float4 over t (coalesced)
            const int b0 = m0 >> 10, t0 = m0 & 1023;
#pragma unroll
            for (int i = 0; i < 4; i++) {
                int idx = i * 128 + tid;
                int mm4 = (idx & 15) * 4, kk = idx >> 4;
                float4 f = *(const float4*)&Afp[(b0 << 20) + ((k0 + kk) << 10) + t0 + mm4];
                As[mm4 + 0][kk] = f2bf(f.x);
                As[mm4 + 1][kk] = f2bf(f.y);
                As[mm4 + 2][kk] = f2bf(f.z);
                As[mm4 + 3][kk] = f2bf(f.w);
            }
        } else if constexpr (MODE == 1 || MODE == 2) {
#pragma unroll
            for (int i = 0; i < 4; i++) {
                int idx = i * 128 + tid;
                int kk4 = (idx & 7) * 4, mm = idx >> 3;
                float4 f = *(const float4*)&Afp[(m0 + mm) * 1024 + k0 + kk4];
                v4bf p;
                p[0] = f2bf(f.x); p[1] = f2bf(f.y); p[2] = f2bf(f.z); p[3] = f2bf(f.w);
                *(v4bf*)&As[mm][kk4] = p;
            }
        } else {
#pragma unroll
            for (int i = 0; i < 2; i++) {
                int idx = i * 128 + tid;
                int kk8 = (idx & 3) * 8, mm = idx >> 2;
                *(uint4*)&As[mm][kk8] = *(const uint4*)&Abf[(m0 + mm) * 1024 + k0 + kk8];
            }
        }
        // Bt[n][k] = W[n][k]; float4 over k (coalesced), packed b64 LDS store
#pragma unroll
        for (int i = 0; i < 4; i++) {
            int idx = i * 128 + tid;
            int kk4 = (idx & 7) * 4, nn = idx >> 3;
            float4 f = *(const float4*)&W[(n0 + nn) * 1024 + k0 + kk4];
            v4bf p;
            p[0] = f2bf(f.x); p[1] = f2bf(f.y); p[2] = f2bf(f.z); p[3] = f2bf(f.w);
            *(v4bf*)&Bt[nn][kk4] = p;
        }
        if (k0 + 32 < 1024)
            __builtin_prefetch(&W[(n0 + (tid >> 1)) * 1024 + k0 + 32], 0, 1);
        __syncthreads();

        // A fragment (16x32 bf16, contiguous k = 16*half + e): lane holds row r; K split by lane half
        v16bf a;
#pragma unroll
        for (int e = 0; e < 16; e++) {
            int v = e >> 1;
            int k = (half << 4) + e;
            a[e] = As[wave * 16 + r][k];
        }
#pragma unroll
        for (int c = 0; c < 4; c++) {
            v16bf bb;
#pragma unroll
            for (int e = 0; e < 16; e++) bb[e] = Bt[c * 16 + r][(half << 4) + e];
            acc[c] = __builtin_amdgcn_wmma_f32_16x16x32_bf16(
                false, a, false, bb, (short)0, acc[c], false, false);
        }
        __syncthreads();
    }

    // ---- epilogue ----
    const int b = m0 >> 10;
    float bia[4];
#pragma unroll
    for (int c = 0; c < 4; c++) bia[c] = bias[n0 + c * 16 + r];

    if constexpr (MODE == 0 || MODE == 1) {
        const float len = lens[b];
        const int h = n0 >> 6;
#pragma unroll
        for (int c = 0; c < 2; c++) {
            int i = c * 16 + r;                      // rotation index 0..31
            float theta = 10.0f * __expf(-(float)i * 0.28782313662425572f); // ln(1e4)/32
#pragma unroll
            for (int rr = 0; rr < 8; rr++) {
                int m = m0 + wave * 16 + rr + (half << 3);
                int t = m & 1023;
                float fr = ((float)t / len) * theta;
                float cs = __cosf(fr), sn = __sinf(fr);
                float v0 = acc[c][rr] + bia[c];
                float v2 = acc[c + 2][rr] + bia[c + 2];
                int base = (((b * 16 + h) << 10) + t) * 64;
                outb[base + i]      = f2bf(v0 * cs - v2 * sn);
                outb[base + i + 32] = f2bf(v0 * sn + v2 * cs);
            }
        }
    } else if constexpr (MODE == 2) {
        const int h = n0 >> 6;
#pragma unroll
        for (int c = 0; c < 4; c++)
#pragma unroll
            for (int rr = 0; rr < 8; rr++) {
                int m = m0 + wave * 16 + rr + (half << 3);
                int t = m & 1023;
                int d = c * 16 + r;
                outb[(((b * 16 + h) << 10) + t) * 64 + d] = f2bf(acc[c][rr] + bia[c]);
            }
    } else {  // MODE 3: out[b, n, t] fp32 with x_mask
#pragma unroll
        for (int c = 0; c < 4; c++)
#pragma unroll
            for (int rr = 0; rr < 8; rr++) {
                int m = m0 + wave * 16 + rr + (half << 3);
                int t = m & 1023;
                int n = n0 + c * 16 + r;
                float vl = (acc[c][rr] + bia[c]) * xmask[(b << 10) + t];
                outf[(b << 20) + (n << 10) + t] = vl;
            }
    }
}

// ---------------------------------------------------------------------------
// Flash attention: grid (T/128, B*H), 256 threads = 8 waves, each wave 16 q-rows.
// Q/K/V: bf16 [B*H][1024][64]. Output O: bf16 [B][T][H*64].
// Ks row-major (S B-frags contiguous); V transposed in LDS (P@V B-frags contiguous).
// K chunk staged with GLOBAL_LOAD_ASYNC_TO_LDS_B128 when the builtin exists.
// ---------------------------------------------------------------------------
__global__ __launch_bounds__(256) void attn_kernel(
    const __bf16* __restrict__ Q, const __bf16* __restrict__ K,
    const __bf16* __restrict__ V, const float* __restrict__ cmask,
    __bf16* __restrict__ O) {
    __shared__ alignas(16) __bf16 Ks[64][80];      // [l][d], 160B row stride
    __shared__ alignas(16) __bf16 Vt[64][80];      // [d][l]
    __shared__ alignas(16) __bf16 Ps[8][16][80];   // per-wave P

    const int tid  = threadIdx.x;
    const int wave = tid >> 5;
    const int lane = tid & 31;
    const int r    = lane & 15;
    const int half = lane >> 4;
    const int bh = blockIdx.y;
    const int b  = bh >> 4;
    const int h  = bh & 15;
    const __bf16* Qb = Q + (bh << 16);
    const __bf16* Kb = K + (bh << 16);
    const __bf16* Vb = V + (bh << 16);
    const int qrow0 = blockIdx.x * 128 + wave * 16;
    const float NEG = -__builtin_inff();

    // Preload this wave's Q fragments (K-dim 64 = 2 WMMA k-steps)
    v16bf aq[2];
#pragma unroll
    for (int st = 0; st < 2; st++)
#pragma unroll
        for (int e = 0; e < 16; e++) {
            int v = e >> 1;
            int k = st * 32 + (half << 4) + e;
            aq[st][e] = Qb[(qrow0 + r) * 64 + k];
        }

    float mrow[8], lsum[8];
    v8f oacc[4] = {};
#pragma unroll
    for (int rr = 0; rr < 8; rr++) { mrow[rr] = NEG; lsum[rr] = 0.0f; }

    for (int l0 = 0; l0 < 1024; l0 += 64) {
        // --- stage K (async LDS DMA if available) ---
#pragma unroll
        for (int i = 0; i < 2; i++) {
            int idx = i * 256 + tid;
            int l = idx >> 3, d8 = (idx & 7) * 8;
            const __bf16* src = &Kb[(l0 + l) * 64 + d8];
            __bf16* dst = &Ks[l][d8];
#ifdef USE_ASYNC_LDS
            __builtin_amdgcn_global_load_async_to_lds_b128(
                (g4p)(unsigned long long)src,
                (l4p)(unsigned)(unsigned long long)dst, 0, 0);
#else
            *(uint4*)dst = *(const uint4*)src;
#endif
        }
        // --- stage V transposed: vector global load, scatter to Vt[d][l] ---
#pragma unroll
        for (int i = 0; i < 2; i++) {
            int idx = i * 256 + tid;
            int l = idx >> 3, d8 = (idx & 7) * 8;
            uint4 raw = *(const uint4*)&Vb[(l0 + l) * 64 + d8];
            const __bf16* e8 = (const __bf16*)&raw;
#pragma unroll
            for (int j = 0; j < 8; j++) Vt[d8 + j][l] = e8[j];
        }
#ifdef USE_ASYNC_LDS
        __builtin_amdgcn_s_wait_asynccnt(0);
#endif
        __syncthreads();

        // S = Q @ K^T  (4 column tiles of 16)
        v8f s4[4] = {};
#pragma unroll
        for (int c = 0; c < 4; c++)
#pragma unroll
            for (int st = 0; st < 2; st++) {
                v16bf bk;
#pragma unroll
                for (int e = 0; e < 16; e++)
                    bk[e] = Ks[c * 16 + r][st * 32 + (half << 4) + e];
                s4[c] = __builtin_amdgcn_wmma_f32_16x16x32_bf16(
                    false, aq[st], false, bk, (short)0, s4[c], false, false);
            }

        // scale 1/sqrt(1024) + context mask
        float cmv[4];
#pragma unroll
        for (int c = 0; c < 4; c++) cmv[c] = cmask[(b << 10) + l0 + c * 16 + r];
        float rowm[8];
#pragma unroll
        for (int rr = 0; rr < 8; rr++) rowm[rr] = NEG;
#pragma unroll
        for (int c = 0; c < 4; c++)
#pragma unroll
            for (int rr = 0; rr < 8; rr++) {
                float x = s4[c][rr] * 0.03125f;
                if (cmv[c] == 0.0f) x = NEG;
                s4[c][rr] = x;
                rowm[rr] = fmaxf(rowm[rr], x);
            }
#pragma unroll
        for (int rr = 0; rr < 8; rr++)
            for (int off = 1; off < 16; off <<= 1)
                rowm[rr] = fmaxf(rowm[rr], __shfl_xor(rowm[rr], off, 32));

        float rs[8];
#pragma unroll
        for (int rr = 0; rr < 8; rr++) {
            float mn = fmaxf(mrow[rr], rowm[rr]);
            float sc = __expf(mrow[rr] - mn);
            lsum[rr] *= sc;
            mrow[rr] = mn;
#pragma unroll
            for (int dt = 0; dt < 4; dt++) oacc[dt][rr] *= sc;
            rs[rr] = 0.0f;
        }
#pragma unroll
        for (int c = 0; c < 4; c++)
#pragma unroll
            for (int rr = 0; rr < 8; rr++) {
                float p = __expf(s4[c][rr] - mrow[rr]);
                rs[rr] += p;
                Ps[wave][rr + (half << 3)][c * 16 + r] = f2bf(p);
            }
#pragma unroll
        for (int rr = 0; rr < 8; rr++) {
            for (int off = 1; off < 16; off <<= 1) rs[rr] += __shfl_xor(rs[rr], off, 32);
            lsum[rr] += rs[rr];
        }

        // O += P @ V  (wave-private P via LDS; DS ops in-order within wave)
#pragma unroll
        for (int st = 0; st < 2; st++) {
            v16bf ap;
#pragma unroll
            for (int e = 0; e < 16; e++) {
                int v = e >> 1;
                int k = st * 32 + (half << 4) + e;
                ap[e] = Ps[wave][r][k];
            }
#pragma unroll
            for (int dt = 0; dt < 4; dt++) {
                v16bf bv;
#pragma unroll
                for (int e = 0; e < 16; e++)
                    bv[e] = Vt[dt * 16 + r][st * 32 + (half << 4) + e];
                oacc[dt] = __builtin_amdgcn_wmma_f32_16x16x32_bf16(
                    false, ap, false, bv, (short)0, oacc[dt], false, false);
            }
        }
        __syncthreads();
    }

    // normalize and store O[b, t, h*64 + d]
#pragma unroll
    for (int dt = 0; dt < 4; dt++)
#pragma unroll
        for (int rr = 0; rr < 8; rr++) {
            int t = qrow0 + rr + (half << 3);
            float ov = oacc[dt][rr] / lsum[rr];
            O[((b << 10) + t) * 1024 + h * 64 + dt * 16 + r] = f2bf(ov);
        }
}

// ---------------------------------------------------------------------------
extern "C" void kernel_launch(void* const* d_in, const int* in_sizes, int n_in,
                              void* d_out, int out_size, void* d_ws, size_t ws_size,
                              hipStream_t stream) {
    const float* x   = (const float*)d_in[0];
    const float* ctx = (const float*)d_in[1];
    const float* xm  = (const float*)d_in[2];
    const float* cm  = (const float*)d_in[3];
    const float* Wq  = (const float*)d_in[4];
    const float* bq  = (const float*)d_in[5];
    const float* Wk  = (const float*)d_in[6];
    const float* bk  = (const float*)d_in[7];
    const float* Wv  = (const float*)d_in[8];
    const float* bv  = (const float*)d_in[9];
    const float* Wo  = (const float*)d_in[10];
    const float* bo  = (const float*)d_in[11];
    float* out = (float*)d_out;

    char* ws = (char*)d_ws;
    float*  lens = (float*)ws;                       // 8 floats (+pad to 256B)
    __bf16* Qw = (__bf16*)(ws + 256);                // [64][1024][64] = 4M bf16
    const size_t QKV = (size_t)4 * 16 * 1024 * 64;
    __bf16* Kw = Qw + QKV;
    __bf16* Vw = Kw + QKV;
    __bf16* Ow = Vw + QKV;                           // [4][1024][1024] bf16

    lens_kernel<<<8, 256, 0, stream>>>(xm, cm, lens);

    dim3 g(64, 16), blk(128);
    gemm_kernel<0><<<g, blk, 0, stream>>>(x,   nullptr, Wq, bq, lens,     nullptr, Qw, nullptr);
    gemm_kernel<1><<<g, blk, 0, stream>>>(ctx, nullptr, Wk, bk, lens + 4, nullptr, Kw, nullptr);
    gemm_kernel<2><<<g, blk, 0, stream>>>(ctx, nullptr, Wv, bv, nullptr,  nullptr, Vw, nullptr);

    attn_kernel<<<dim3(8, 64), 256, 0, stream>>>(Qw, Kw, Vw, cm, Ow);

    gemm_kernel<3><<<g, blk, 0, stream>>>(nullptr, Ow, Wo, bo, nullptr, xm, nullptr, out);
}
